// Decoder_87187836109109
// MI455X (gfx1250) — matrix operand order repaired and checked by reading, not compiled, not run on hardware
//
#include <hip/hip_runtime.h>
#include <hip/hip_bf16.h>

// LSTM decoder: B=512, H=512, T=1024, out_dim=1.
// Folded recurrence: gates = h @ W_eff^T + b_eff with
//   W_eff = W_hh + W_ih @ W_fc   ([4H,H]),  b_eff = b_ih+b_hh + b_fc*W_ih
// (step 0 uses plain W_hh / b_ih+b_hh since x0 == 0).
// Per step: one fused WMMA GEMM + LSTM cell kernel. y_{s-1} is computed by
// step s from its *input* h; a tail kernel emits y_{T-1}.
//
// Step kernel geometry: grid (16,16), block 64 (2 wave32s) -> 256 WGs, 512 waves.
// Wave tile: 32 rows x 16 cols of hn = 2 M-subtiles x 4 gates = 8 accumulators.
// Each B fragment (W_eff row segment) is reused by 2 WMMAs -> 1.5 loads/WMMA.

#define B_DIM 512
#define H_DIM 512
#define FH    2048   // 4*H
#define T_LEN 1024

typedef __attribute__((ext_vector_type(16))) _Float16 v16h;
typedef __attribute__((ext_vector_type(8)))  _Float16 v8h;
typedef __attribute__((ext_vector_type(8)))  float    v8f;

union V16U { v16h v; v8h h[2]; };

__device__ __forceinline__ float sigmoidf_(float x) {
  return 1.0f / (1.0f + __expf(-x));
}

// ---------------- prep kernels ----------------

__global__ void k_prep_weights(const float* __restrict__ W_hh,
                               const float* __restrict__ W_ih,
                               const float* __restrict__ W_fc,
                               _Float16* __restrict__ W0,
                               _Float16* __restrict__ We) {
  int idx = blockIdx.x * blockDim.x + threadIdx.x;
  if (idx >= FH * H_DIM) return;
  int n = idx >> 9;          // row in [0,4H)
  int k = idx & (H_DIM - 1); // col in [0,H)
  float w = W_hh[idx];
  W0[idx] = (_Float16)w;
  We[idx] = (_Float16)(w + W_ih[n] * W_fc[k]);
}

__global__ void k_prep_biases(const float* __restrict__ b_ih,
                              const float* __restrict__ b_hh,
                              const float* __restrict__ W_ih,
                              const float* __restrict__ b_fc,
                              float* __restrict__ b0,
                              float* __restrict__ be) {
  int i = blockIdx.x * blockDim.x + threadIdx.x;
  if (i >= FH) return;
  float v = b_ih[i] + b_hh[i];
  b0[i] = v;
  be[i] = v + b_fc[0] * W_ih[i];
}

__global__ void k_init_state(const float* __restrict__ h0,
                             const float* __restrict__ c0,
                             _Float16* __restrict__ hA,
                             float* __restrict__ cst) {
  int i = blockIdx.x * blockDim.x + threadIdx.x;
  if (i >= B_DIM * H_DIM) return;
  hA[i]  = (_Float16)h0[i];
  cst[i] = c0[i];
}

// ---------------- fused LSTM step ----------------

__global__ __launch_bounds__(64)
void k_lstm_step(const _Float16* __restrict__ h_in,
                 _Float16* __restrict__ h_out,
                 float* __restrict__ c_st,
                 const _Float16* __restrict__ W,    // [4H][H] row-major f16
                 const float* __restrict__ bias,    // [4H]
                 const float* __restrict__ W_fc,    // [H]
                 const float* __restrict__ b_fc,    // [1]
                 float* __restrict__ out,           // [B][T]
                 int s) {
  const int tid    = threadIdx.x;
  const int lane   = tid & 31;
  const int wave   = tid >> 5;     // 0..1
  const int tile_m = blockIdx.x;   // 0..15 -> 32 rows each
  const int tile_n = blockIdx.y;   // 0..15 -> 32 cols each

  // Side job: y_{s-1} = h_in @ W_fc^T + b_fc (h_in == hn_{s-1}).
  // Distributed: WG (tile_m, tile_m) handles its own 32 rows (16 per wave).
  // Branch is wave-uniform, so EXEC stays all-1s for WMMA.
  if (tile_n == (tile_m & 15) && s > 0) {
    #pragma unroll 4
    for (int j = 0; j < 16; ++j) {
      int row = tile_m * 32 + wave * 16 + j;
      const _Float16* hr = h_in + (size_t)row * H_DIM + lane * 16;
      const float*    wr = W_fc + lane * 16;
      float sum = 0.f;
      #pragma unroll
      for (int e = 0; e < 16; ++e) sum += (float)hr[e] * wr[e];
      #pragma unroll
      for (int off = 16; off >= 1; off >>= 1) sum += __shfl_xor(sum, off, 32);
      if (lane == 0) out[(size_t)row * T_LEN + (s - 1)] = sum + b_fc[0];
    }
  }

  const int m_base = tile_m * 32;              // wave covers rows [m_base, m_base+32)
  const int n_base = tile_n * 32 + wave * 16;  // 16 cols per wave

  const int lane_lo = lane & 15;
  const int lane_hi = lane >> 4;   // 0/1

  // A fragments (16x32 f16): lanes 0-15 hold row m=lane, K 0..15;
  // lanes 16-31 hold row m=lane-16, K 16..31 (matches B). Two contiguous 16B loads each.
  const _Float16* a_row0 = h_in + (size_t)(m_base +      lane_lo) * H_DIM + (lane_hi << 4);
  const _Float16* a_row1 = h_in + (size_t)(m_base + 16 + lane_lo) * H_DIM + (lane_hi << 4);

  // B fragments (32x16): lanes 0-15 hold col n, K 0..15; lanes 16-31 K 16..31.
  // W stored row-major [n][k] == column-major of B -> one contiguous 32B load.
  const _Float16* b_row0 = W + (size_t)(0 * H_DIM + n_base + lane_lo) * H_DIM + (lane_hi << 4);
  const _Float16* b_row1 = W + (size_t)(1 * H_DIM + n_base + lane_lo) * H_DIM + (lane_hi << 4);
  const _Float16* b_row2 = W + (size_t)(2 * H_DIM + n_base + lane_lo) * H_DIM + (lane_hi << 4);
  const _Float16* b_row3 = W + (size_t)(3 * H_DIM + n_base + lane_lo) * H_DIM + (lane_hi << 4);

  v8f acc[4][2];
  #pragma unroll
  for (int g = 0; g < 4; ++g) { acc[g][0] = (v8f){}; acc[g][1] = (v8f){}; }

  #pragma unroll 2
  for (int kk = 0; kk < H_DIM; kk += 32) {
    V16U a0, a1;
    a0.h[0] = *(const v8h*)(a_row0 + kk);
    a0.h[1] = *(const v8h*)(a_row0 + kk + 8);
    a1.h[0] = *(const v8h*)(a_row1 + kk);
    a1.h[1] = *(const v8h*)(a_row1 + kk + 8);
    __builtin_prefetch(a_row0 + kk + 32, 0, 3);  // global_prefetch_b8 (near-temporal)
    v16h b0 = *(const v16h*)(b_row0 + kk);
    v16h b1 = *(const v16h*)(b_row1 + kk);
    v16h b2 = *(const v16h*)(b_row2 + kk);
    v16h b3 = *(const v16h*)(b_row3 + kk);
    acc[0][0] = __builtin_amdgcn_wmma_f32_16x16x32_f16(false, a0.v, false, b0, (short)0, acc[0][0], false, false);
    acc[0][1] = __builtin_amdgcn_wmma_f32_16x16x32_f16(false, a1.v, false, b0, (short)0, acc[0][1], false, false);
    acc[1][0] = __builtin_amdgcn_wmma_f32_16x16x32_f16(false, a0.v, false, b1, (short)0, acc[1][0], false, false);
    acc[1][1] = __builtin_amdgcn_wmma_f32_16x16x32_f16(false, a1.v, false, b1, (short)0, acc[1][1], false, false);
    acc[2][0] = __builtin_amdgcn_wmma_f32_16x16x32_f16(false, a0.v, false, b2, (short)0, acc[2][0], false, false);
    acc[2][1] = __builtin_amdgcn_wmma_f32_16x16x32_f16(false, a1.v, false, b2, (short)0, acc[2][1], false, false);
    acc[3][0] = __builtin_amdgcn_wmma_f32_16x16x32_f16(false, a0.v, false, b3, (short)0, acc[3][0], false, false);
    acc[3][1] = __builtin_amdgcn_wmma_f32_16x16x32_f16(false, a1.v, false, b3, (short)0, acc[3][1], false, false);
  }

  // C/D layout: lane 0-15 -> N=lane, VGPR r -> M=r; lanes 16-31 -> N=lane-16, M=r+8.
  const int   n_col = n_base + lane_lo;
  const float bi = bias[0 * H_DIM + n_col];
  const float bf = bias[1 * H_DIM + n_col];
  const float bg = bias[2 * H_DIM + n_col];
  const float bo = bias[3 * H_DIM + n_col];

  #pragma unroll
  for (int hlf = 0; hlf < 2; ++hlf) {
    #pragma unroll
    for (int r = 0; r < 8; ++r) {
      const int    m   = m_base + hlf * 16 + r + (lane_hi << 3);
      const size_t off = (size_t)m * H_DIM + n_col;
      float iv = sigmoidf_(acc[0][hlf][r] + bi);
      float fv = sigmoidf_(acc[1][hlf][r] + bf);
      float gv = tanhf(acc[2][hlf][r] + bg);
      float ov = sigmoidf_(acc[3][hlf][r] + bo);
      float cn = fv * c_st[off] + iv * gv;
      c_st[off]  = cn;
      h_out[off] = (_Float16)(ov * tanhf(cn));
    }
  }
}

// ---------------- tail: y_{T-1} ----------------

__global__ void k_final_y(const _Float16* __restrict__ h_in,
                          const float* __restrict__ W_fc,
                          const float* __restrict__ b_fc,
                          float* __restrict__ out) {
  int lane = threadIdx.x & 31;
  int wave = threadIdx.x >> 5;
  int row  = blockIdx.x * 8 + wave;
  const _Float16* hr = h_in + (size_t)row * H_DIM + lane * 16;
  const float*    wr = W_fc + lane * 16;
  float sum = 0.f;
  #pragma unroll
  for (int e = 0; e < 16; ++e) sum += (float)hr[e] * wr[e];
  #pragma unroll
  for (int off = 16; off >= 1; off >>= 1) sum += __shfl_xor(sum, off, 32);
  if (lane == 0) out[(size_t)row * T_LEN + (T_LEN - 1)] = sum + b_fc[0];
}

// ---------------- host ----------------

extern "C" void kernel_launch(void* const* d_in, const int* in_sizes, int n_in,
                              void* d_out, int out_size, void* d_ws, size_t ws_size,
                              hipStream_t stream) {
  (void)in_sizes; (void)n_in; (void)out_size; (void)ws_size;
  const float* h0   = (const float*)d_in[0];
  const float* c0   = (const float*)d_in[1];
  const float* W_ih = (const float*)d_in[2];
  const float* W_hh = (const float*)d_in[3];
  const float* b_ih = (const float*)d_in[4];
  const float* b_hh = (const float*)d_in[5];
  const float* W_fc = (const float*)d_in[6];
  const float* b_fc = (const float*)d_in[7];
  float* out = (float*)d_out;

  char* ws = (char*)d_ws;                       // all offsets 256B-aligned
  _Float16* W0  = (_Float16*)(ws + 0);          // 2 MB  [4H][H] f16 (step 0)
  _Float16* We  = (_Float16*)(ws + 2097152);    // 2 MB  W_eff f16
  float*    b0  = (float*)   (ws + 4194304);    // 8 KB
  float*    be  = (float*)   (ws + 4202496);    // 8 KB
  _Float16* hA  = (_Float16*)(ws + 4210688);    // 512 KB h ping
  _Float16* hB  = (_Float16*)(ws + 4734976);    // 512 KB h pong
  float*    cst = (float*)   (ws + 5259264);    // 1 MB  c state (f32)

  k_prep_weights<<<(FH * H_DIM + 255) / 256, 256, 0, stream>>>(W_hh, W_ih, W_fc, W0, We);
  k_prep_biases <<<(FH + 255) / 256,         256, 0, stream>>>(b_ih, b_hh, W_ih, b_fc, b0, be);
  k_init_state  <<<(B_DIM * H_DIM + 255) / 256, 256, 0, stream>>>(h0, c0, hA, cst);

  for (int s = 0; s < T_LEN; ++s) {
    const _Float16* Wp   = (s == 0) ? W0 : We;
    const float*    bp   = (s == 0) ? b0 : be;
    const _Float16* hin  = (s & 1) ? hB : hA;
    _Float16*       hout = (s & 1) ? hA : hB;
    k_lstm_step<<<dim3(16, 16), 64, 0, stream>>>(hin, hout, cst, Wp, bp, W_fc, b_fc, out, s);
  }
  // T even -> final h is in hA
  k_final_y<<<64, 256, 0, stream>>>(hA, W_fc, b_fc, out);
}
